// MultiHeadAttention_4964982194830
// MI455X (gfx1250) — matrix operand order repaired and checked by reading, not compiled, not run on hardware
//
#include <hip/hip_runtime.h>
#include <hip/hip_bf16.h>

#define EMB   1024
#define HEADS 16
#define HD    64
#define BATCH 4
#define SEQ   2048

typedef __attribute__((ext_vector_type(16))) __bf16 v16bf;
typedef __attribute__((ext_vector_type(8)))  __bf16 v8bf;
typedef __attribute__((ext_vector_type(8)))  float  v8f;

__device__ __forceinline__ unsigned short f32_to_bf16(float f) {
    union { float f; unsigned int u; } c; c.f = f;
    unsigned int u = c.u;
    u += 0x7FFFu + ((u >> 16) & 1u);   // round-to-nearest-even
    return (unsigned short)(u >> 16);
}

__device__ __forceinline__ v16bf combine16(v8bf lo, v8bf hi) {
    return __builtin_shufflevector(lo, hi, 0,1,2,3,4,5,6,7,8,9,10,11,12,13,14,15);
}

__device__ __forceinline__ v8f wmma_bf16(v16bf a, v16bf b, v8f c) {
    return __builtin_amdgcn_wmma_f32_16x16x32_bf16(false, a, false, b, (short)0, c, false, false);
}

// ---------------------------------------------------------------- convert
__global__ __launch_bounds__(256)
void cvt_f32_bf16(const float* __restrict__ src, unsigned short* __restrict__ dst, int n4) {
    int i = (blockIdx.x * 256 + threadIdx.x);
    if (i >= n4) return;
    float4 f = ((const float4*)src)[i];
    ushort4 o;
    o.x = f32_to_bf16(f.x); o.y = f32_to_bf16(f.y);
    o.z = f32_to_bf16(f.z); o.w = f32_to_bf16(f.w);
    ((ushort4*)dst)[i] = o;
}

// ---------------------------------------------------------------- GEMM: out = A[M,K](bf16) * W[N,K]^T + bias
// One wave -> 32x64 output tile (2 A-fragments share 4 B-fragments per K-step,
// 8 WMMAs per iteration). 8 waves / block.
template<bool BF16_OUT>
__global__ __launch_bounds__(256)
void gemm_wmma(const unsigned short* __restrict__ A,
               const unsigned short* __restrict__ W,
               const float* __restrict__ bias,
               void* __restrict__ out,
               int M, int N, int K) {
    const int lane = threadIdx.x & 31;
    const int wave = threadIdx.x >> 5;
    const int gw   = blockIdx.x * 8 + wave;
    const int tilesN = N >> 6;
    const int wm = gw / tilesN;
    const int wn = gw % tilesN;
    if (wm * 32 >= M) return;
    const int row0 = wm * 32;
    const int col0 = wn * 64;
    const int h = lane >> 4;   // half of wave
    const int r = lane & 15;

    v8f acc[2][4] = {{v8f{}, v8f{}, v8f{}, v8f{}}, {v8f{}, v8f{}, v8f{}, v8f{}}};

    const unsigned short* arow0 = A + (size_t)(row0 + r) * K;
    const unsigned short* arow1 = arow0 + (size_t)16 * K;
    const unsigned short* wbase = W + (size_t)col0 * K;

    for (int k0 = 0; k0 < K; k0 += 32) {
        __builtin_prefetch(arow0 + k0 + 128, 0, 0);
        __builtin_prefetch(arow1 + k0 + 128, 0, 0);
        // A fragments: lane(h,r): M=r, e[0..7]=K k0+8h.., e[8..15]=K k0+8h+16..
        v16bf af0 = combine16(*(const v8bf*)(arow0 + k0 + 8 * h),
                              *(const v8bf*)(arow0 + k0 + 8 * h + 16));
        v16bf af1 = combine16(*(const v8bf*)(arow1 + k0 + 8 * h),
                              *(const v8bf*)(arow1 + k0 + 8 * h + 16));
        #pragma unroll
        for (int t = 0; t < 4; ++t) {
            // B fragment: column n = col0+16t+r -> row n of W, e[0..7]=K k0+8h.., e[8..15]=K k0+8h+16..
            const unsigned short* wrow = wbase + (size_t)(16 * t + r) * K + k0 + 8 * h;
            v16bf bfrag = combine16(*(const v8bf*)(wrow), *(const v8bf*)(wrow + 16));
            acc[0][t] = wmma_bf16(af0, bfrag, acc[0][t]);
            acc[1][t] = wmma_bf16(af1, bfrag, acc[1][t]);
        }
    }

    // epilogue: lane(h,r) set s, vgpr j -> row row0+16s+8h+j, col col0+16t+r
    #pragma unroll
    for (int s = 0; s < 2; ++s) {
        #pragma unroll
        for (int t = 0; t < 4; ++t) {
            const int col = col0 + 16 * t + r;
            const float bv = bias[col];
            #pragma unroll
            for (int j = 0; j < 8; ++j) {
                const float v = acc[s][t][j] + bv;
                const size_t idx = (size_t)(row0 + 16 * s + 8 * h + j) * N + col;
                if (BF16_OUT) ((unsigned short*)out)[idx] = f32_to_bf16(v);
                else          ((float*)out)[idx] = v;
            }
        }
    }
}

// ---------------------------------------------------------------- V transpose: [B,S,E] -> [B,H,HD,S]
__global__ __launch_bounds__(256)
void transpose_v(const unsigned short* __restrict__ Vp, unsigned short* __restrict__ Vt) {
    int idx = blockIdx.x * 256 + threadIdx.x;
    if (idx >= BATCH * SEQ * EMB) return;
    int e = idx % EMB;
    int s = (idx / EMB) % SEQ;
    int b = idx / (EMB * SEQ);
    int hh = e / HD, d = e % HD;
    Vt[(((size_t)b * HEADS + hh) * HD + d) * SEQ + s] = Vp[idx];
}

// ---------------------------------------------------------------- flash attention
// One wave -> 16 query rows of one (b,h). 8 waves / block.
__global__ __launch_bounds__(256)
void attn_wmma(const unsigned short* __restrict__ Qp,
               const unsigned short* __restrict__ Kp,
               const unsigned short* __restrict__ Vt,
               unsigned short* __restrict__ Out) {
    __shared__ unsigned short plds[8 * 16 * 32];   // per-wave 16x32 P staging
    const int lane = threadIdx.x & 31;
    const int wave = threadIdx.x >> 5;
    const int gw   = blockIdx.x * 8 + wave;
    const int QB = SEQ / 16;
    if (gw >= BATCH * HEADS * QB) return;
    const int b   = gw / (HEADS * QB);
    const int rem = gw % (HEADS * QB);
    const int hh  = rem / QB;
    const int q0  = (rem % QB) * 16;
    const int h = lane >> 4;
    const int r = lane & 15;
    unsigned short* myp = plds + wave * (16 * 32);

    const float C = 0.125f * 1.44269504088896340736f;  // 1/sqrt(HD) * log2(e)

    // Q fragments for d0=0 and d0=32
    const unsigned short* qrow = Qp + (size_t)(b * SEQ + q0 + r) * EMB + hh * HD;
    v16bf qf0 = combine16(*(const v8bf*)(qrow + 8 * h),      *(const v8bf*)(qrow + 8 * h + 16));
    v16bf qf1 = combine16(*(const v8bf*)(qrow + 32 + 8 * h), *(const v8bf*)(qrow + 32 + 8 * h + 16));

    v8f o[4] = {v8f{}, v8f{}, v8f{}, v8f{}};
    float m[8], l[8];
    #pragma unroll
    for (int j = 0; j < 8; ++j) { m[j] = -3.0e38f; l[j] = 0.0f; }

    for (int jb = 0; jb < SEQ; jb += 32) {
        // ---- scores: two 16x16 tiles over keys [jb,jb+16) and [jb+16,jb+32)
        v8f sa = {}, sb = {};
        {
            const unsigned short* krow = Kp + (size_t)(b * SEQ + jb + r) * EMB + hh * HD;
            v16bf kf0 = combine16(*(const v8bf*)(krow + 8 * h), *(const v8bf*)(krow + 8 * h + 16));
            v16bf kf1 = combine16(*(const v8bf*)(krow + 32 + 8 * h), *(const v8bf*)(krow + 32 + 8 * h + 16));
            sa = wmma_bf16(qf0, kf0, sa);
            sa = wmma_bf16(qf1, kf1, sa);
            const unsigned short* krow2 = krow + (size_t)16 * EMB;
            v16bf kg0 = combine16(*(const v8bf*)(krow2 + 8 * h), *(const v8bf*)(krow2 + 8 * h + 16));
            v16bf kg1 = combine16(*(const v8bf*)(krow2 + 32 + 8 * h), *(const v8bf*)(krow2 + 32 + 8 * h + 16));
            sb = wmma_bf16(qf0, kg0, sb);
            sb = wmma_bf16(qf1, kg1, sb);
        }
        // ---- online softmax per query row (row j lives in one VGPR across a 16-lane half)
        #pragma unroll
        for (int j = 0; j < 8; ++j) {
            float bm = fmaxf(sa[j], sb[j]);
            bm = fmaxf(bm, __shfl_xor(bm, 1, 32));
            bm = fmaxf(bm, __shfl_xor(bm, 2, 32));
            bm = fmaxf(bm, __shfl_xor(bm, 4, 32));
            bm = fmaxf(bm, __shfl_xor(bm, 8, 32));
            const float mn   = fmaxf(m[j], bm);
            const float corr = __builtin_amdgcn_exp2f((m[j] - mn) * C);
            const float pa   = __builtin_amdgcn_exp2f((sa[j] - mn) * C);
            const float pb   = __builtin_amdgcn_exp2f((sb[j] - mn) * C);
            float bs = pa + pb;
            bs += __shfl_xor(bs, 1, 32);
            bs += __shfl_xor(bs, 2, 32);
            bs += __shfl_xor(bs, 4, 32);
            bs += __shfl_xor(bs, 8, 32);
            l[j] = l[j] * corr + bs;
            m[j] = mn;
            o[0][j] *= corr; o[1][j] *= corr; o[2][j] *= corr; o[3][j] *= corr;
            // stage P (accumulator layout -> row-major 16x32 tile in LDS)
            myp[(8 * h + j) * 32 + r]      = f32_to_bf16(pa);
            myp[(8 * h + j) * 32 + 16 + r] = f32_to_bf16(pb);
        }
        asm volatile("" ::: "memory");
        // ---- reload P as an A-fragment (16 queries x 32 keys)
        v16bf pf = combine16(*(const v8bf*)(myp + r * 32 + 8 * h),
                             *(const v8bf*)(myp + r * 32 + 8 * h + 16));
        // ---- P x V  (V^T rows are contiguous over keys)
        const unsigned short* vbase = Vt + ((size_t)(b * HEADS + hh) * HD) * SEQ + jb + 8 * h;
        #pragma unroll
        for (int t = 0; t < 4; ++t) {
            const unsigned short* vrow = vbase + (size_t)(16 * t + r) * SEQ;
            v16bf vf = combine16(*(const v8bf*)(vrow), *(const v8bf*)(vrow + 16));
            o[t] = wmma_bf16(pf, vf, o[t]);
        }
        asm volatile("" ::: "memory");
    }

    // ---- normalize and store bf16 [B,S,E]
    #pragma unroll
    for (int j = 0; j < 8; ++j) {
        const float inv = 1.0f / l[j];
        const size_t rowbase = (size_t)(b * SEQ + q0 + 8 * h + j) * EMB + hh * HD;
        #pragma unroll
        for (int t = 0; t < 4; ++t)
            Out[rowbase + 16 * t + r] = f32_to_bf16(o[t][j] * inv);
    }
}

// ---------------------------------------------------------------- host launcher
extern "C" void kernel_launch(void* const* d_in, const int* in_sizes, int n_in,
                              void* d_out, int out_size, void* d_ws, size_t ws_size,
                              hipStream_t stream) {
    const float* q  = (const float*)d_in[0];
    const float* k  = (const float*)d_in[1];
    const float* v  = (const float*)d_in[2];
    const float* Wq = (const float*)d_in[3];
    const float* bq = (const float*)d_in[4];
    const float* Wk = (const float*)d_in[5];
    const float* bk = (const float*)d_in[6];
    const float* Wv = (const float*)d_in[7];
    const float* bv = (const float*)d_in[8];
    const float* Wo = (const float*)d_in[9];
    const float* bo = (const float*)d_in[10];

    const size_t NQ = (size_t)BATCH * SEQ * EMB;   // 8,388,608
    const size_t NW = (size_t)EMB * EMB;           // 1,048,576

    unsigned short* p   = (unsigned short*)d_ws;
    unsigned short* qb  = p; p += NQ;
    unsigned short* kb  = p; p += NQ;
    unsigned short* vb  = p; p += NQ;
    unsigned short* wqb = p; p += NW;
    unsigned short* wkb = p; p += NW;
    unsigned short* wvb = p; p += NW;
    unsigned short* wob = p; p += NW;
    unsigned short* Qp  = p; p += NQ;
    unsigned short* Kp  = p; p += NQ;
    unsigned short* Vp  = p; p += NQ;
    unsigned short* Vt  = p; p += NQ;
    unsigned short* At  = p; p += NQ;

    const int gQ = (int)(NQ / 4 / 256);   // convert grids (exact multiples)
    const int gW = (int)(NW / 4 / 256);
    cvt_f32_bf16<<<gQ, 256, 0, stream>>>(q,  qb,  (int)(NQ / 4));
    cvt_f32_bf16<<<gQ, 256, 0, stream>>>(k,  kb,  (int)(NQ / 4));
    cvt_f32_bf16<<<gQ, 256, 0, stream>>>(v,  vb,  (int)(NQ / 4));
    cvt_f32_bf16<<<gW, 256, 0, stream>>>(Wq, wqb, (int)(NW / 4));
    cvt_f32_bf16<<<gW, 256, 0, stream>>>(Wk, wkb, (int)(NW / 4));
    cvt_f32_bf16<<<gW, 256, 0, stream>>>(Wv, wvb, (int)(NW / 4));
    cvt_f32_bf16<<<gW, 256, 0, stream>>>(Wo, wob, (int)(NW / 4));

    const int M = BATCH * SEQ;  // 8192
    // waves = (M/32)*(EMB/64) = 256*16 = 4096 -> 512 blocks of 8 waves
    const int gemmBlocks = (M / 32) * (EMB / 64) / 8;
    gemm_wmma<true><<<gemmBlocks, 256, 0, stream>>>(qb, wqb, bq, Qp, M, EMB, EMB);
    gemm_wmma<true><<<gemmBlocks, 256, 0, stream>>>(kb, wkb, bk, Kp, M, EMB, EMB);
    gemm_wmma<true><<<gemmBlocks, 256, 0, stream>>>(vb, wvb, bv, Vp, M, EMB, EMB);

    transpose_v<<<(int)(NQ / 256), 256, 0, stream>>>(Vp, Vt);

    // waves = B*H*(S/16) = 8192 -> 1024 blocks
    const int attnBlocks = BATCH * HEADS * (SEQ / 16) / 8;
    attn_wmma<<<attnBlocks, 256, 0, stream>>>(Qp, Kp, Vt, At);

    gemm_wmma<false><<<gemmBlocks, 256, 0, stream>>>(At, wob, bo, (float*)d_out, M, EMB, EMB);
}
